// CausalfAttention_89515708383662
// MI455X (gfx1250) — matrix operand order repaired and checked by reading, not compiled, not run on hardware
//
#include <hip/hip_runtime.h>

#define BATCH  4
#define SEQ    2048
#define DMODEL 1024
#define SCALE  (1.0f / 32.0f)   // 1/sqrt(d_k), d_k = 1024

typedef __attribute__((ext_vector_type(16))) __bf16 v16bf;
typedef __attribute__((ext_vector_type(8)))  __bf16 v8bf;
typedef __attribute__((ext_vector_type(4)))  __bf16 v4bf;
typedef __attribute__((ext_vector_type(8)))  float  v8f;
typedef __attribute__((ext_vector_type(4)))  int    v4i;

// ---------------------------------------------------------------------------
// CDNA5 async global->LDS copy (ASYNCcnt-tracked) with fallbacks.
// Builtin signature (per hipcc diagnostic): first arg is v4i addrspace(1)*.
// ---------------------------------------------------------------------------
typedef __attribute__((address_space(1))) v4i gv4i_t;   // global 128-bit chunk
typedef __attribute__((address_space(3))) v4i lv4i_t;   // LDS    128-bit chunk

#if __has_builtin(__builtin_amdgcn_s_wait_asynccnt)
#define WAIT_ASYNC(n) __builtin_amdgcn_s_wait_asynccnt(n)
#else
#define WAIT_ASYNC(n) asm volatile("s_wait_asynccnt " #n ::: "memory")
#endif

__device__ __forceinline__ void async_ld16(const __bf16* g, __bf16* l) {
#if __has_builtin(__builtin_amdgcn_global_load_async_to_lds_b128)
  __builtin_amdgcn_global_load_async_to_lds_b128((gv4i_t*)g, (lv4i_t*)l, 0, 0);
#else
  unsigned loff = (unsigned)(size_t)(__attribute__((address_space(3))) __bf16*)l;
  unsigned long long ga = (unsigned long long)(size_t)g;
  asm volatile("global_load_async_to_lds_b128 %0, %1, off"
               :: "v"(loff), "v"(ga) : "memory");
#endif
}

// ---------------------------------------------------------------------------
// WMMA fragment helpers (CDNA5 v_wmma_f32_16x16x32_bf16, wave32)
// ---------------------------------------------------------------------------
__device__ __forceinline__ v16bf frag_cat(v8bf lo, v8bf hi) {
  return __builtin_shufflevector(lo, hi, 0, 1, 2, 3, 4, 5, 6, 7,
                                         8, 9, 10, 11, 12, 13, 14, 15);
}

// A fragment: 16x32 (MxK) bf16 from row-major [M x ld], K contiguous.
__device__ __forceinline__ v16bf load_fragA(const __bf16* base, long ld, int lane) {
  const int m  = lane & 15;
  const int kh = lane >> 4;
  const __bf16* p = base + (long)m * ld + kh * 8;
  v8bf lo = *(const v8bf*)(p);
  v8bf hi = *(const v8bf*)(p + 16);
  return frag_cat(lo, hi);
}

// B fragment: 32x16 (KxN) bf16 sourced from row-major [N x ld] (=> A*B^T GEMM).
__device__ __forceinline__ v16bf load_fragB(const __bf16* base, long ld, int lane) {
  const int n  = lane & 15;
  const int kh = lane >> 4;
  const __bf16* p = base + (long)n * ld + kh * 8;
  v8bf lo = *(const v8bf*)(p);
  v8bf hi = *(const v8bf*)(p + 16);
  return frag_cat(lo, hi);
}

__device__ __forceinline__ v8f wmma_bf16(v16bf a, v16bf b, v8f c) {
  return __builtin_amdgcn_wmma_f32_16x16x32_bf16(false, a, false, b,
                                                 (short)0, c, false, false);
}

#define LDS_PITCH 40
#define TILE_E (128 * LDS_PITCH)   // one 128x32 bf16 tile, 40-elem row pitch

// Cooperative async load of a 128x32 bf16 tile (row-major, stride ld) into LDS.
// 256 threads x 2 async b128 each.
__device__ __forceinline__ void async_tile_load(__bf16* lds, const __bf16* src,
                                                long ld, int tid) {
  const int row = tid >> 1;           // 0..127
  const int kh  = tid & 1;            // 16-element half
  const __bf16* s = src + (long)row * ld + kh * 16;
  __bf16* d = lds + row * LDS_PITCH + kh * 16;
  async_ld16(s, d);
  async_ld16(s + 8, d + 8);
}

// ---------------------------------------------------------------------------
// Triple-buffered A*B^T inner loop: one s_barrier per 32-wide K step.
// Step i:  wait(own tile-i async loads) -> barrier (tile i ready everywhere,
// everyone done reading tile i-1) -> issue tile i+2 into buf (i+2)%3 (that
// buffer was last read at step i-1) -> 8 WMMAs on tile i.
// ---------------------------------------------------------------------------
__device__ __forceinline__ void gemm_loop(v8f acc[8],
                                          const __bf16* aw, long a_ld,
                                          const __bf16* bsrc, long b_ld,
                                          int kend, __bf16* lds,
                                          int tid, int lane) {
  const int N = kend >> 5;
  __syncthreads();                       // prior users of LDS are done
  async_tile_load(lds, bsrc, b_ld, tid);
  if (N > 1) async_tile_load(lds + TILE_E, bsrc + 32, b_ld, tid);
  for (int i = 0; i < N; ++i) {
    if (i + 1 < N) { WAIT_ASYNC(2); } else { WAIT_ASYNC(0); }
    __syncthreads();
    if (i + 2 < N)
      async_tile_load(lds + ((i + 2) % 3) * TILE_E,
                      bsrc + (long)(i + 2) * 32, b_ld, tid);
    v16bf a = load_fragA(aw + (long)i * 32, a_ld, lane);
    __bf16* lt = lds + (i % 3) * TILE_E;
#pragma unroll
    for (int j = 0; j < 8; ++j)
      acc[j] = wmma_bf16(a, load_fragB(lt + j * 16 * LDS_PITCH, LDS_PITCH, lane),
                         acc[j]);
  }
}

__device__ __forceinline__ void zero_acc(v8f acc[8]) {
#pragma unroll
  for (int j = 0; j < 8; ++j)
#pragma unroll
    for (int i = 0; i < 8; ++i) acc[j][i] = 0.0f;
}

// ---------------------------------------------------------------------------
// K0: f32 -> bf16 conversion of x and the three weight matrices
// ---------------------------------------------------------------------------
__global__ __launch_bounds__(256) void convert_kernel(
    const float* __restrict__ x,  const float* __restrict__ wq,
    const float* __restrict__ wk, const float* __restrict__ wv,
    __bf16* __restrict__ xh, __bf16* __restrict__ wqh,
    __bf16* __restrict__ wkh, __bf16* __restrict__ wvh) {
  const long NX4 = (long)BATCH * SEQ * DMODEL / 4;  // 2,097,152
  const long NW4 = (long)DMODEL * DMODEL / 4;       //   262,144
  long i = (long)blockIdx.x * 256 + threadIdx.x;
  const float* src;
  __bf16* dst;
  long j;
  if (i < NX4)                { src = x;  dst = xh;  j = i; }
  else if (i < NX4 + NW4)     { src = wq; dst = wqh; j = i - NX4; }
  else if (i < NX4 + 2 * NW4) { src = wk; dst = wkh; j = i - NX4 - NW4; }
  else                        { src = wv; dst = wvh; j = i - NX4 - 2 * NW4; }
  float4 f = ((const float4*)src)[j];
  v4bf o;
  o[0] = (__bf16)f.x; o[1] = (__bf16)f.y; o[2] = (__bf16)f.z; o[3] = (__bf16)f.w;
  ((v4bf*)dst)[j] = o;
}

// ---------------------------------------------------------------------------
// K1: QKV projections.  Q,K stored [b,s,e]; V stored transposed [b,e,s].
// 128x128 tile per block, 8 waves, each wave a 16x128 strip.
// ---------------------------------------------------------------------------
__global__ __launch_bounds__(256) void qkv_kernel(
    const __bf16* __restrict__ xh,  const __bf16* __restrict__ wqh,
    const __bf16* __restrict__ wkh, const __bf16* __restrict__ wvh,
    __bf16* __restrict__ Qh, __bf16* __restrict__ Kh, __bf16* __restrict__ Vth) {
  __shared__ __bf16 lds[3 * TILE_E];
  const int tid = threadIdx.x, wid = tid >> 5, lane = tid & 31;
  const int b = blockIdx.z, s0 = blockIdx.x * 128, e0 = blockIdx.y * 128;
  const __bf16* xw = xh + ((long)b * SEQ + s0 + wid * 16) * DMODEL;
  const __bf16* Ws[3] = {wqh, wkh, wvh};
  const int col = lane & 15, rbase = (lane >> 4) * 8;

#pragma unroll
  for (int m = 0; m < 3; ++m) {
    v8f acc[8];
    zero_acc(acc);
    gemm_loop(acc, xw, DMODEL, Ws[m] + (long)e0 * DMODEL, DMODEL, DMODEL,
              lds, tid, lane);
    if (m < 2) {
      __bf16* dst = (m == 0) ? Qh : Kh;
#pragma unroll
      for (int j = 0; j < 8; ++j)
#pragma unroll
        for (int i = 0; i < 8; ++i) {
          const int srow = s0 + wid * 16 + rbase + i;
          const int ecol = e0 + j * 16 + col;
          dst[((long)b * SEQ + srow) * DMODEL + ecol] = (__bf16)acc[j][i];
        }
    } else {
      // V^T store: each lane's 8 values are s-consecutive -> one b128 store/tile
#pragma unroll
      for (int j = 0; j < 8; ++j) {
        v8bf pk;
#pragma unroll
        for (int i = 0; i < 8; ++i) pk[i] = (__bf16)acc[j][i];
        const int e = e0 + j * 16 + col;
        *(v8bf*)(Vth + ((long)b * DMODEL + e) * SEQ + s0 + wid * 16 + rbase) = pk;
      }
    }
  }
}

// ---------------------------------------------------------------------------
// K2: scores = Q @ K^T (f32).  Skip blocks above the diagonal; store k <= q.
// ---------------------------------------------------------------------------
__global__ __launch_bounds__(256) void scores_kernel(
    const __bf16* __restrict__ Qh, const __bf16* __restrict__ Kh,
    float* __restrict__ Sc) {
  __shared__ __bf16 lds[3 * TILE_E];
  const int tid = threadIdx.x, wid = tid >> 5, lane = tid & 31;
  const int b = blockIdx.z, q0 = blockIdx.y * 128, k0 = blockIdx.x * 128;
  if (k0 > q0) return;  // block-uniform: fully masked tile
  const __bf16* qw = Qh + ((long)b * SEQ + q0 + wid * 16) * DMODEL;

  v8f acc[8];
  zero_acc(acc);
  gemm_loop(acc, qw, DMODEL, Kh + ((long)b * SEQ + k0) * DMODEL, DMODEL, DMODEL,
            lds, tid, lane);

  const int col = lane & 15, rbase = (lane >> 4) * 8;
#pragma unroll
  for (int j = 0; j < 8; ++j)
#pragma unroll
    for (int i = 0; i < 8; ++i) {
      const int q = q0 + wid * 16 + rbase + i;
      const int k = k0 + j * 16 + col;
      if (k <= q) Sc[((long)b * SEQ + q) * SEQ + k] = acc[j][i];
    }
}

// ---------------------------------------------------------------------------
// K3: row softmax, p = exp((s - m) * SCALE) / l (bf16 out); zero-fill up to
// the 128-boundary so the PV GEMM can run dense K tiles to the diagonal.
// ---------------------------------------------------------------------------
__global__ __launch_bounds__(256) void softmax_kernel(
    const float* __restrict__ Sc, __bf16* __restrict__ Ph) {
  const int row = blockIdx.x;             // b * SEQ + q
  const int q   = row & (SEQ - 1);
  const int tid = threadIdx.x;
  const float* srow = Sc + (long)row * SEQ;
  __bf16* prow = Ph + (long)row * SEQ;
  const int n = q + 1;
  __shared__ float red[256];

  float m = -__builtin_inff();
  for (int k = tid; k < n; k += 256) m = fmaxf(m, srow[k]);
  red[tid] = m;
  __syncthreads();
  for (int s = 128; s > 0; s >>= 1) {
    if (tid < s) red[tid] = fmaxf(red[tid], red[tid + s]);
    __syncthreads();
  }
  m = red[0];
  __syncthreads();

  float l = 0.0f;
  for (int k = tid; k < n; k += 256) l += __expf((srow[k] - m) * SCALE);
  red[tid] = l;
  __syncthreads();
  for (int s = 128; s > 0; s >>= 1) {
    if (tid < s) red[tid] += red[tid + s];
    __syncthreads();
  }
  l = red[0];
  const float inv = 1.0f / l;

  const int nend = (n + 127) & ~127;
  for (int k = tid; k < nend; k += 256) {
    float p = (k < n) ? __expf((srow[k] - m) * SCALE) * inv : 0.0f;
    prow[k] = (__bf16)p;
  }
}

// ---------------------------------------------------------------------------
// K4: out = P @ V via V^T [b,e,s] (A*B^T form).  K loop bounded at q0+128.
// ---------------------------------------------------------------------------
__global__ __launch_bounds__(256) void pv_kernel(
    const __bf16* __restrict__ Ph, const __bf16* __restrict__ Vth,
    float* __restrict__ out) {
  __shared__ __bf16 lds[3 * TILE_E];
  const int tid = threadIdx.x, wid = tid >> 5, lane = tid & 31;
  const int b = blockIdx.z, q0 = blockIdx.x * 128, e0 = blockIdx.y * 128;
  const int kend = q0 + 128;
  const __bf16* pw = Ph + ((long)b * SEQ + q0 + wid * 16) * SEQ;

  v8f acc[8];
  zero_acc(acc);
  gemm_loop(acc, pw, SEQ, Vth + ((long)b * DMODEL + e0) * SEQ, SEQ, kend,
            lds, tid, lane);

  const int col = lane & 15, rbase = (lane >> 4) * 8;
#pragma unroll
  for (int j = 0; j < 8; ++j)
#pragma unroll
    for (int i = 0; i < 8; ++i) {
      const int q = q0 + wid * 16 + rbase + i;
      const int e = e0 + j * 16 + col;
      out[((long)b * SEQ + q) * DMODEL + e] = acc[j][i];
    }
}

// ---------------------------------------------------------------------------
// Workspace layout (bytes), total 174,063,616:
//   xh@0 (16M)  Wq/Wk/Wv h@16M (3x2M)  Qh@23,068,672  Kh@39,845,888
//   Vth@56,623,104  Sc(f32)@73,400,320 (64M)  Ph@140,509,184 (32M)
// ---------------------------------------------------------------------------
extern "C" void kernel_launch(void* const* d_in, const int* in_sizes, int n_in,
                              void* d_out, int out_size, void* d_ws, size_t ws_size,
                              hipStream_t stream) {
  const float* x  = (const float*)d_in[0];
  const float* Wq = (const float*)d_in[1];
  const float* Wk = (const float*)d_in[2];
  const float* Wv = (const float*)d_in[3];
  float* out = (float*)d_out;

  char* ws = (char*)d_ws;
  __bf16* xh  = (__bf16*)(ws + 0);
  __bf16* wqh = (__bf16*)(ws + 16777216);
  __bf16* wkh = (__bf16*)(ws + 18874368);
  __bf16* wvh = (__bf16*)(ws + 20971520);
  __bf16* Qh  = (__bf16*)(ws + 23068672);
  __bf16* Kh  = (__bf16*)(ws + 39845888);
  __bf16* Vth = (__bf16*)(ws + 56623104);
  float*  Sc  = (float*) (ws + 73400320);
  __bf16* Ph  = (__bf16*)(ws + 140509184);

  convert_kernel<<<11264, 256, 0, stream>>>(x, Wq, Wk, Wv, xh, wqh, wkh, wvh);
  qkv_kernel<<<dim3(SEQ / 128, DMODEL / 128, BATCH), 256, 0, stream>>>(
      xh, wqh, wkh, wvh, Qh, Kh, Vth);
  scores_kernel<<<dim3(SEQ / 128, SEQ / 128, BATCH), 256, 0, stream>>>(Qh, Kh, Sc);
  softmax_kernel<<<dim3(BATCH * SEQ), 256, 0, stream>>>(Sc, Ph);
  pv_kernel<<<dim3(SEQ / 128, DMODEL / 128, BATCH), 256, 0, stream>>>(Ph, Vth, out);
}
